// HL_HGCNN_65266323030051
// MI455X (gfx1250) — matrix operand order repaired and checked by reading, not compiled, not run on hardware
//
#include <hip/hip_runtime.h>

// ---------------------------------------------------------------------------
// HL-HGCNN forward for MI455X (gfx1250, wave32).
// GEMMs: v_wmma_f32_16x16x32_f16, f16 operands / f32 accumulate.
//  - all GEMM A-operands (activations) stored f16 -> pure b128 fragment loads
//  - weights pre-converted to f16 column-major (32B per B fragment)
//  - 4 output tiles per wave (shared A fragment, 4 WMMAs per K-step)
// Scatter ops use global_atomic_add_f32 via unsafeAtomicAdd.
// ---------------------------------------------------------------------------

typedef __attribute__((ext_vector_type(16))) _Float16 v16h;
typedef __attribute__((ext_vector_type(8)))  _Float16 v8h;
typedef __attribute__((ext_vector_type(8)))  float    v8f;

#define TPB 256

static __host__ __device__ inline int cdiv_i(long a, long b) { return (int)((a + b - 1) / b); }

// ---------------- scatter / gather / elementwise kernels -------------------

// out[idx0[e], c] += w[e] * x[idx1[e], c]   (f32 accumulate)
__global__ void k_spmm(const int* __restrict__ idx, const float* __restrict__ w,
                       const float* __restrict__ x, float* __restrict__ out,
                       int nE, int F, int ldx, int ldo) {
  long i = (long)blockIdx.x * blockDim.x + threadIdx.x;
  if (i >= (long)nE * F) return;
  int e = (int)(i / F), c = (int)(i - (long)e * F);
  int d = idx[e];
  int s = idx[nE + e];
  unsafeAtomicAdd(out + (long)d * ldo + c, w[e] * x[(long)s * ldx + c]);
}

// z[u[e],c] -= xs[e,c]; z[v[e],c] += xs[e,c]   (par1_mm; f16 source)
__global__ void k_par1(const int* __restrict__ ei, const _Float16* __restrict__ xs,
                       float* __restrict__ z, int E, int F, int ldx, int ldz) {
  long i = (long)blockIdx.x * blockDim.x + threadIdx.x;
  if (i >= (long)E * F) return;
  int e = (int)(i / F), c = (int)(i - (long)e * F);
  float v = (float)xs[(long)e * ldx + c];
  unsafeAtomicAdd(z + (long)ei[e] * ldz + c, -v);
  unsafeAtomicAdd(z + (long)ei[E + e] * ldz + c, v);
}

// t2s[e,c] = 0.5*|xt[v[e],c] - xt[u[e],c]|   (par1T_mm; f16 in/out)
__global__ void k_t2s(const int* __restrict__ ei, const _Float16* __restrict__ xt,
                      _Float16* __restrict__ o, int E, int F, int ldx, int ldo) {
  long i = (long)blockIdx.x * blockDim.x + threadIdx.x;
  if (i >= (long)E * F) return;
  int e = (int)(i / F), c = (int)(i - (long)e * F);
  int u = ei[e], v = ei[E + e];
  float d = (float)xt[(long)v * ldx + c] - (float)xt[(long)u * ldx + c];
  o[(long)e * ldo + c] = (_Float16)(0.5f * fabsf(d));
}

__global__ void k_deg(const int* __restrict__ ei, float* __restrict__ D, int n) {
  long i = (long)blockIdx.x * blockDim.x + threadIdx.x;
  if (i < n) unsafeAtomicAdd(D + ei[i], 1.0f);
}

__global__ void k_eps(float* __restrict__ D, int n) {
  long i = (long)blockIdx.x * blockDim.x + threadIdx.x;
  if (i < n) D[i] += 1e-6f;
}

// zh = (f16)(|z| / D[row])
__global__ void k_absdiv(const float* __restrict__ z, const float* __restrict__ D,
                         _Float16* __restrict__ zh, int M, int F) {
  long i = (long)blockIdx.x * blockDim.x + threadIdx.x;
  if (i >= (long)M * F) return;
  int r = (int)(i / F);
  zh[i] = (_Float16)(fabsf(z[i]) / D[r]);
}

// init-stage: lx = x - lx (f32 in place, distinct leading dims)
__global__ void k_sub_f32(float* __restrict__ lx, const float* __restrict__ x,
                          int M, int F, int ldx, int ldl) {
  long i = (long)blockIdx.x * blockDim.x + threadIdx.x;
  if (i >= (long)M * F) return;
  int r = (int)(i / F), c = (int)(i - (long)r * F);
  long li = (long)r * ldl + c;
  lx[li] = x[(long)r * ldx + c] - lx[li];
}

// block-stage: outh = (f16)(x - lx)   (GEMM-operand producer)
__global__ void k_sub_h(const float* __restrict__ lx, const float* __restrict__ x,
                        _Float16* __restrict__ outh, long n) {
  long i = (long)blockIdx.x * blockDim.x + threadIdx.x;
  if (i >= n) return;
  outh[i] = (_Float16)(x[i] - lx[i]);
}

// a *= sigmoid(g); also emit f16 copy for the following GEMM
__global__ void k_gate(float* __restrict__ a, const float* __restrict__ g,
                       _Float16* __restrict__ ah, long n) {
  long i = (long)blockIdx.x * blockDim.x + threadIdx.x;
  if (i >= n) return;
  float v = a[i] * (1.0f / (1.0f + expf(-g[i])));
  a[i] = v;
  ah[i] = (_Float16)v;
}

// W (K x N, f32 row-major) -> Wt (N x K, f16) : Wt[n*K+k] = W[k*N+n]
__global__ void k_cvt_wt(const float* __restrict__ W, _Float16* __restrict__ Wt,
                         int K, int N) {
  long i = (long)blockIdx.x * blockDim.x + threadIdx.x;
  if (i >= (long)K * N) return;
  int n = (int)(i / K), k = (int)(i - (long)n * K);
  Wt[i] = (_Float16)W[(long)k * N + n];
}

// tiny-K GEMM for the init stage (K = 1 or 2)
__global__ void k_small_lag(const float* __restrict__ x, int ldx,
                            const float* __restrict__ bm, int ldb,
                            const float* __restrict__ W0, const float* __restrict__ W1,
                            const float* __restrict__ bias, float* __restrict__ y,
                            int M, int K, int fo) {
  long i = (long)blockIdx.x * blockDim.x + threadIdx.x;
  if (i >= (long)M * fo) return;
  int r = (int)(i / fo), c = (int)(i - (long)r * fo);
  float acc = bias[c];
  for (int k = 0; k < K; ++k)
    acc += x[(long)r * ldx + k] * W0[k * fo + c] + bm[(long)r * ldb + k] * W1[k * fo + c];
  y[i] = acc;
}

// per-column mean / inv-std over M rows
__global__ void k_bn_stats(const float* __restrict__ y, int M, int F,
                           float* __restrict__ mean, float* __restrict__ inv) {
  const int c = blockIdx.x;
  __shared__ float s1[TPB], s2[TPB];
  float a = 0.0f, b = 0.0f;
  for (int r = threadIdx.x; r < M; r += TPB) {
    float v = y[(long)r * F + c];
    a += v; b += v * v;
  }
  s1[threadIdx.x] = a; s2[threadIdx.x] = b;
  __syncthreads();
  for (int s = TPB / 2; s > 0; s >>= 1) {
    if ((int)threadIdx.x < s) { s1[threadIdx.x] += s1[threadIdx.x + s];
                                s2[threadIdx.x] += s2[threadIdx.x + s]; }
    __syncthreads();
  }
  if (threadIdx.x == 0) {
    float m = s1[0] / (float)M;
    float v = s2[0] / (float)M - m * m;
    mean[c] = m;
    inv[c]  = rsqrtf(v + 1e-5f);
  }
}

// out[:, co:co+F] = relu((y - mean)*inv*g + b) -> f16 concat buffer
__global__ void k_bn_relu(const float* __restrict__ y, const float* __restrict__ mean,
                          const float* __restrict__ inv, const float* __restrict__ g,
                          const float* __restrict__ bb, _Float16* __restrict__ out,
                          int M, int F, int ldo, int co) {
  long i = (long)blockIdx.x * blockDim.x + threadIdx.x;
  if (i >= (long)M * F) return;
  int r = (int)(i / F), c = (int)(i - (long)r * F);
  float v = (y[i] - mean[c]) * inv[c] * g[c] + bb[c];
  out[(long)r * ldo + co + c] = (_Float16)fmaxf(v, 0.0f);
}

// ---------------- WMMA GEMM ------------------------------------------------
// C[M,N] = A0[M,K] @ W0 (+ A1[M,K] @ W1) + bias[N]; A f16 row-major,
// weights f16 column-major (Wt[n*K+k]). One wave computes four adjacent
// 16x16 C tiles (cols n0..n0+63) sharing the A fragment: per 32-wide K-step
// -> 2 b128 A loads + 4x 32B B loads + 4 WMMAs, zero converts.
// Requires: M mult 16, N mult 64, K mult 32 (guaranteed by this model).
// Fragment layouts used here (contiguous k order, same for A and B):
//   A 16x32 f16 : lane half h, row m0+(lane&15); elem j -> K = 16h + j
//                 (two adjacent 16B runs)
//   B 32x16 f16 : col n0+(lane&15); elem j -> K = 16h + j (contiguous in Wt)
//   C/D 16x16 f32: elem r -> row m0 + 8h + r, col n0+(lane&15)
__global__ __launch_bounds__(256) void k_gemm_wmma(
    const _Float16* __restrict__ A0, int lda0, const _Float16* __restrict__ Wt0,
    const _Float16* __restrict__ A1, int lda1, const _Float16* __restrict__ Wt1,
    const float* __restrict__ bias, float* __restrict__ C,
    int M, int N, int K, int ldc) {
  const int lane = threadIdx.x & 31;
  const int wave = threadIdx.x >> 5;
  const int m0 = (blockIdx.x * 8 + wave) * 16;
  if (m0 >= M) return;                       // wave-uniform: EXEC stays all-ones
  const int n0 = blockIdx.y * 64;
  const int half = lane >> 4;
  const int mr = m0 + (lane & 15);
  const int nc = n0 + (lane & 15);

  v8f acc0 = {}, acc1 = {}, acc2 = {}, acc3 = {};
  for (int pass = 0; pass < 2; ++pass) {
    const _Float16* A = pass ? A1 : A0;
    if (!A) continue;
    const _Float16* Wt = pass ? Wt1 : Wt0;
    const int lda = pass ? lda1 : lda0;
    const _Float16* arow = A + (long)mr * lda + half * 16;
    const _Float16* w0 = Wt + (long)(nc)      * K + half * 16;
    const _Float16* w1 = Wt + (long)(nc + 16) * K + half * 16;
    const _Float16* w2 = Wt + (long)(nc + 32) * K + half * 16;
    const _Float16* w3 = Wt + (long)(nc + 48) * K + half * 16;
    for (int k = 0; k < K; k += 32) {
      const v8h lo = *(const v8h*)(arow + k);
      const v8h hi = *(const v8h*)(arow + k + 8);
      const v16h a = __builtin_shufflevector(lo, hi, 0, 1, 2, 3, 4, 5, 6, 7,
                                             8, 9, 10, 11, 12, 13, 14, 15);
      const v16h b0 = *(const v16h*)(w0 + k);
      const v16h b1 = *(const v16h*)(w1 + k);
      const v16h b2 = *(const v16h*)(w2 + k);
      const v16h b3 = *(const v16h*)(w3 + k);
      acc0 = __builtin_amdgcn_wmma_f32_16x16x32_f16(false, a, false, b0, (short)0, acc0, false, false);
      acc1 = __builtin_amdgcn_wmma_f32_16x16x32_f16(false, a, false, b1, (short)0, acc1, false, false);
      acc2 = __builtin_amdgcn_wmma_f32_16x16x32_f16(false, a, false, b2, (short)0, acc2, false, false);
      acc3 = __builtin_amdgcn_wmma_f32_16x16x32_f16(false, a, false, b3, (short)0, acc3, false, false);
    }
  }
  const float bv0 = bias ? bias[nc]      : 0.0f;
  const float bv1 = bias ? bias[nc + 16] : 0.0f;
  const float bv2 = bias ? bias[nc + 32] : 0.0f;
  const float bv3 = bias ? bias[nc + 48] : 0.0f;
  float* crow = C + (long)(m0 + half * 8) * ldc + nc;
#pragma unroll
  for (int r = 0; r < 8; ++r) {
    crow[(long)r * ldc]      = acc0[r] + bv0;
    crow[(long)r * ldc + 16] = acc1[r] + bv1;
    crow[(long)r * ldc + 32] = acc2[r] + bv2;
    crow[(long)r * ldc + 48] = acc3[r] + bv3;
  }
}

// ---------------- final projection -----------------------------------------
// out[e] = (dot(xs_last[e], W[0:256]) + dot(|xt[v]-xt[u]|/2, W[256:512]) + b) * mask[e]
__global__ void k_final(const _Float16* __restrict__ xs0h, const _Float16* __restrict__ xt0h,
                        const int* __restrict__ ei, const float* __restrict__ W,
                        const float* __restrict__ bias, const float* __restrict__ x_s,
                        float* __restrict__ out, int E) {
  int e = blockIdx.x * 8 + (threadIdx.x >> 5);
  if (e >= E) return;
  int lane = threadIdx.x & 31;
  int u = ei[e], v = ei[E + e];
  const _Float16* xrow = xs0h + (long)e * 960 + 704;
  const _Float16* xu = xt0h + (long)u * 960 + 704;
  const _Float16* xv = xt0h + (long)v * 960 + 704;
  float acc = 0.0f;
  for (int c = lane; c < 256; c += 32) {
    acc += (float)xrow[c] * W[c];
    acc += 0.5f * fabsf((float)xv[c] - (float)xu[c]) * W[256 + c];
  }
  for (int off = 16; off > 0; off >>= 1) acc += __shfl_xor(acc, off, 32);
  if (lane == 0) out[e] = (acc + bias[0]) * x_s[2 * e + 1];
}

// ---------------- host orchestration ---------------------------------------

// params flatten order (JAX pytree: dict keys sorted):
enum { AS_W = 0, AS_B, AT_W, AT_B, BNS_B, BNS_G, BNT_B, BNT_G,
       CS_W0, CS_W1, CS_B, CT_W0, CT_W1, CT_B, LS_W, LS_B, LT_W, LT_B };

extern "C" void kernel_launch(void* const* d_in, const int* in_sizes, int n_in,
                              void* d_out, int out_size, void* d_ws, size_t ws_size,
                              hipStream_t stream) {
  const int N = 30000, E = 60000, NT = 150000, NS = 240000;
  const int OUTS[6] = {64, 64, 128, 128, 256, 256};
  const int FI[6]   = {64, 128, 192, 320, 448, 704};

  const float* x_t = (const float*)d_in[0];
  const float* x_s = (const float*)d_in[1];
  const int* ei    = (const int*)d_in[2];  // (2,E)
  const int* ei_t  = (const int*)d_in[3];  // (2,NT)
  const float* w_t = (const float*)d_in[4];
  const int* ei_s  = (const int*)d_in[5];  // (2,NS)
  const float* w_s = (const float*)d_in[6];
  auto BPt = [&](int b, int which) { return (const float*)d_in[7 + b * 18 + which]; };
  const int IB = 7 + 6 * 18;  // init params: Ws0,Ws1,Wt0,Wt1,bbs,bbt,bs,bt,gs,gt
  const float* Ws0 = (const float*)d_in[IB + 0];
  const float* Ws1 = (const float*)d_in[IB + 1];
  const float* Wt0 = (const float*)d_in[IB + 2];
  const float* Wt1 = (const float*)d_in[IB + 3];
  const float* bbs = (const float*)d_in[IB + 4];
  const float* bbt = (const float*)d_in[IB + 5];
  const float* bs  = (const float*)d_in[IB + 6];
  const float* bt  = (const float*)d_in[IB + 7];
  const float* gs  = (const float*)d_in[IB + 8];
  const float* gt  = (const float*)d_in[IB + 9];
  const float* outW = (const float*)d_in[IB + 10];
  const float* outB = (const float*)d_in[IB + 11];
  float* out = (float*)d_out;

  // workspace carve-up: f32 region first, then f16 region
  float* ws = (float*)d_ws;
  size_t o = 0;
  float* s2t  = ws + o; o += (size_t)N * 704;   // f32 atomic accumulator
  float* tmp1 = ws + o; o += (size_t)E * 256;
  float* tmp2 = ws + o; o += (size_t)E * 256;
  float* lxb  = ws + o; o += (size_t)E * 256;
  float* Dd   = ws + o; o += (size_t)N;
  float* mnb  = ws + o; o += 256;
  float* ivb  = ws + o; o += 320;               // pad to 64-elt alignment
  _Float16* hb = (_Float16*)(ws + o);
  size_t ho = 0;
  _Float16* xt0h  = hb + ho; ho += (size_t)N * 960;  // concat node feats (f16)
  _Float16* xs0h  = hb + ho; ho += (size_t)E * 960;  // concat edge feats (f16)
  _Float16* s2th  = hb + ho; ho += (size_t)N * 704;
  _Float16* t2sh  = hb + ho; ho += (size_t)E * 704;
  _Float16* tmp1h = hb + ho; ho += (size_t)E * 256;
  _Float16* lxh   = hb + ho; ho += (size_t)E * 256;
  _Float16* wt_cur = hb + ho;                        // f16 transposed-weight arena
  (void)ws_size; (void)in_sizes; (void)n_in; (void)out_size;

  auto EW = [](long n) { return dim3(cdiv_i(n, TPB)); };

  // pre-convert every GEMM weight to f16 column-major
  auto cvtW = [&](const float* W, int K, int Nn) {
    _Float16* dst = wt_cur;
    wt_cur += (size_t)K * Nn;
    k_cvt_wt<<<EW((long)K * Nn), TPB, 0, stream>>>(W, dst, K, Nn);
    return (const _Float16*)dst;
  };
  const _Float16 *ltW[6], *atW[6], *lsW[6], *asW[6], *ct0W[6], *ct1W[6], *cs0W[6], *cs1W[6];
  for (int b = 0; b < 6; ++b) {
    const int fi = FI[b], fo = OUTS[b];
    ltW[b]  = cvtW(BPt(b, LT_W),  fi, fo);
    atW[b]  = cvtW(BPt(b, AT_W),  fi, fo);
    lsW[b]  = cvtW(BPt(b, LS_W),  fi, fo);
    asW[b]  = cvtW(BPt(b, AS_W),  fi, fo);
    ct0W[b] = cvtW(BPt(b, CT_W0), fo, fo);
    ct1W[b] = cvtW(BPt(b, CT_W1), fo, fo);
    cs0W[b] = cvtW(BPt(b, CS_W0), fo, fo);
    cs1W[b] = cvtW(BPt(b, CS_W1), fo, fo);
  }

  auto gemm = [&](const _Float16* A0, int lda0, const _Float16* W0,
                  const _Float16* A1, int lda1, const _Float16* W1,
                  const float* bias, float* C, int M, int Nn, int K, int ldc) {
    dim3 g(cdiv_i(M / 16, 8), Nn / 64);
    k_gemm_wmma<<<g, TPB, 0, stream>>>(A0, lda0, W0, A1, lda1, W1, bias, C, M, Nn, K, ldc);
  };

  // ---- init: node stream (K=2) ----
  hipMemsetAsync(lxb, 0, (size_t)N * 2 * sizeof(float), stream);
  k_spmm<<<EW((long)NT * 2), TPB, 0, stream>>>(ei_t, w_t, x_t, lxb, NT, 2, 2, 2);
  k_sub_f32<<<EW((long)N * 2), TPB, 0, stream>>>(lxb, x_t, N, 2, 2, 2);
  k_small_lag<<<EW((long)N * 64), TPB, 0, stream>>>(x_t, 2, lxb, 2, Wt0, Wt1, bt, tmp1, N, 2, 64);
  k_bn_stats<<<64, TPB, 0, stream>>>(tmp1, N, 64, mnb, ivb);
  k_bn_relu<<<EW((long)N * 64), TPB, 0, stream>>>(tmp1, mnb, ivb, gt, bbt, xt0h, N, 64, 960, 0);

  // ---- init: edge stream (K=1, feat = x_s[:,0]) ----
  hipMemsetAsync(lxb, 0, (size_t)E * sizeof(float), stream);
  k_spmm<<<EW((long)NS), TPB, 0, stream>>>(ei_s, w_s, x_s, lxb, NS, 1, 2, 1);
  k_sub_f32<<<EW((long)E), TPB, 0, stream>>>(lxb, x_s, E, 1, 2, 1);
  k_small_lag<<<EW((long)E * 64), TPB, 0, stream>>>(x_s, 2, lxb, 1, Ws0, Ws1, bs, tmp1, E, 1, 64);
  k_bn_stats<<<64, TPB, 0, stream>>>(tmp1, E, 64, mnb, ivb);
  k_bn_relu<<<EW((long)E * 64), TPB, 0, stream>>>(tmp1, mnb, ivb, gs, bbs, xs0h, E, 64, 960, 0);

  // ---- degree ----
  hipMemsetAsync(Dd, 0, (size_t)N * sizeof(float), stream);
  k_deg<<<EW((long)2 * E), TPB, 0, stream>>>(ei, Dd, 2 * E);
  k_eps<<<EW((long)N), TPB, 0, stream>>>(Dd, N);

  // ---- blocks ----
  for (int b = 0; b < 6; ++b) {
    const int fi = FI[b], fo = OUTS[b], co = FI[b];

    // s2t = |par1_mm(xs0[:, :fi])| / D   (f32 atomics, then f16 for GEMM)
    hipMemsetAsync(s2t, 0, (size_t)N * fi * sizeof(float), stream);
    k_par1<<<EW((long)E * fi), TPB, 0, stream>>>(ei, xs0h, s2t, E, fi, 960, fi);
    k_absdiv<<<EW((long)N * fi), TPB, 0, stream>>>(s2t, Dd, s2th, N, fi);
    // t2s = |par1T_mm(xt0[:, :fi])| / 2   (f16)
    k_t2s<<<EW((long)E * fi), TPB, 0, stream>>>(ei, xt0h, t2sh, E, fi, 960, fi);

    // --- node stream: gated linear then lag-conv + BN + ReLU ---
    gemm(xt0h, 960, ltW[b], nullptr, 0, nullptr, BPt(b, LT_B), tmp1, N, fo, fi, fo);
    gemm(s2th, fi,  atW[b], nullptr, 0, nullptr, BPt(b, AT_B), tmp2, N, fo, fi, fo);
    k_gate<<<EW((long)N * fo), TPB, 0, stream>>>(tmp1, tmp2, tmp1h, (long)N * fo);
    hipMemsetAsync(lxb, 0, (size_t)N * fo * sizeof(float), stream);
    k_spmm<<<EW((long)NT * fo), TPB, 0, stream>>>(ei_t, w_t, tmp1, lxb, NT, fo, fo, fo);
    k_sub_h<<<EW((long)N * fo), TPB, 0, stream>>>(lxb, tmp1, lxh, (long)N * fo);
    gemm(tmp1h, fo, ct0W[b], lxh, fo, ct1W[b], BPt(b, CT_B), tmp2, N, fo, fo, fo);
    k_bn_stats<<<fo, TPB, 0, stream>>>(tmp2, N, fo, mnb, ivb);
    k_bn_relu<<<EW((long)N * fo), TPB, 0, stream>>>(tmp2, mnb, ivb, BPt(b, BNT_G), BPt(b, BNT_B),
                                                    xt0h, N, fo, 960, co);

    // --- edge stream ---
    gemm(xs0h, 960, lsW[b], nullptr, 0, nullptr, BPt(b, LS_B), tmp1, E, fo, fi, fo);
    gemm(t2sh, fi,  asW[b], nullptr, 0, nullptr, BPt(b, AS_B), tmp2, E, fo, fi, fo);
    k_gate<<<EW((long)E * fo), TPB, 0, stream>>>(tmp1, tmp2, tmp1h, (long)E * fo);
    hipMemsetAsync(lxb, 0, (size_t)E * fo * sizeof(float), stream);
    k_spmm<<<EW((long)NS * fo), TPB, 0, stream>>>(ei_s, w_s, tmp1, lxb, NS, fo, fo, fo);
    k_sub_h<<<EW((long)E * fo), TPB, 0, stream>>>(lxb, tmp1, lxh, (long)E * fo);
    gemm(tmp1h, fo, cs0W[b], lxh, fo, cs1W[b], BPt(b, CS_B), tmp2, E, fo, fo, fo);
    k_bn_stats<<<fo, TPB, 0, stream>>>(tmp2, E, fo, mnb, ivb);
    k_bn_relu<<<EW((long)E * fo), TPB, 0, stream>>>(tmp2, mnb, ivb, BPt(b, BNS_G), BPt(b, BNS_B),
                                                    xs0h, E, fo, 960, co);
  }

  // ---- final: inline t2s gather + 512->1 projection ----
  k_final<<<cdiv_i(E, 8), TPB, 0, stream>>>(xs0h, xt0h, ei, outW, outB, x_s, out, E);
}
